// Affinity_propagation_65541200937406
// MI455X (gfx1250) — compile-verified, matrix operand order checked by reading, not run on hardware
//
#include <hip/hip_runtime.h>
#include <hip/hip_bf16.h>

// ---------------------------------------------------------------------------
// Affinity propagation disparity refinement, MI455X (gfx1250, wave32, WMMA).
// Convs as implicit GEMM on v_wmma_f32_16x16x32_f16. Activations fp16 NHWC
// (channels padded to 16) -> every A-fragment lane load is one 32B global
// load. Packed weights staged in LDS per workgroup (ds_load path). Interior
// tiles take a wave-uniform guard-free path (no per-lane EXEC churn).
// ---------------------------------------------------------------------------

typedef __attribute__((ext_vector_type(16))) _Float16 v16h;
typedef __attribute__((ext_vector_type(8)))  float    v8f;

#define NBATCH 8
#define HH 256
#define WW 256
#define HL 512
#define WL 512
#define NPIX (NBATCH * HH * WW)
#define BN_EPS 1e-5f

#define W1P_HALFS (5 * 2 * 32 * 16)   // 5120  (10 KB)
#define W2P_HALFS (9 * 4 * 32 * 16)   // 18432 (36 KB)
#define W3P_HALFS (2 * 32 * 16)       // 1024  (2 KB)

// ---------------------------------------------------------------------------
// Kernel 0: fused resize(2x2 avg) + LR warp error + concat -> fp16 NHWC16.
// Channels: 0-2 normal, 3-5 cur_l, 6-8 cur_r, 9-11 disp_err, 12-15 zero pad.
// ---------------------------------------------------------------------------
__global__ __launch_bounds__(256) void prep_feats_kernel(
    const float* __restrict__ normal,   // [N,3,H,W]
    const float* __restrict__ left,     // [N,3,512,512]
    const float* __restrict__ right,    // [N,3,512,512]
    const float* __restrict__ disp,     // [N,1,H,W]
    _Float16* __restrict__ feats)       // [N,H,W,16]
{
    int idx = blockIdx.x * blockDim.x + threadIdx.x;
    if (idx >= NPIX) return;
    int n   = idx / (HH * WW);
    int rem = idx - n * HH * WW;
    int y   = rem / WW;
    int x   = rem - y * WW;

    // 512->256 half-pixel bilinear == exact 2x2 average (weights fixed 0.5).
    float cl[3], cr[3];
#pragma unroll
    for (int c = 0; c < 3; ++c) {
        const float* Lp = left  + ((((size_t)n * 3 + c) * HL + 2 * y) * WL + 2 * x);
        const float* Rp = right + ((((size_t)n * 3 + c) * HL + 2 * y) * WL + 2 * x);
        cl[c] = 0.25f * (Lp[0] + Lp[1] + Lp[WL] + Lp[WL + 1]);
        cr[c] = 0.25f * (Rp[0] + Rp[1] + Rp[WL] + Rp[WL + 1]);
    }

    float d   = disp[idx];
    float xs  = (float)x - d;
    float x0f = floorf(xs);
    float fr  = xs - x0f;
    int   x0i = (int)x0f;
    int   x1i = x0i + 1;
    bool  vb0 = (x0i >= 0) && (x0i < WW);
    bool  vb1 = (x1i >= 0) && (x1i < WW);
    int   x0c = min(max(x0i, 0), WW - 1);
    int   x1c = min(max(x1i, 0), WW - 1);

    float de[3];
#pragma unroll
    for (int c = 0; c < 3; ++c) {
        const float* Rrow = right + (((size_t)n * 3 + c) * HL + 2 * y) * WL;
        float r0 = 0.25f * (Rrow[2 * x0c] + Rrow[2 * x0c + 1] +
                            Rrow[WL + 2 * x0c] + Rrow[WL + 2 * x0c + 1]);
        float r1 = 0.25f * (Rrow[2 * x1c] + Rrow[2 * x1c + 1] +
                            Rrow[WL + 2 * x1c] + Rrow[WL + 2 * x1c + 1]);
        float warped = (vb0 ? r0 : 0.f) * (1.f - fr) + (vb1 ? r1 : 0.f) * fr;
        de[c] = fabsf(cl[c] - warped);
    }

    __align__(32) _Float16 tmp[16];
#pragma unroll
    for (int c = 0; c < 3; ++c) {
        tmp[c]     = (_Float16)normal[(((size_t)n * 3 + c) * HH + y) * WW + x];
        tmp[3 + c] = (_Float16)cl[c];
        tmp[6 + c] = (_Float16)cr[c];
        tmp[9 + c] = (_Float16)de[c];
    }
#pragma unroll
    for (int c = 12; c < 16; ++c) tmp[c] = (_Float16)0.f;
    *reinterpret_cast<v16h*>(feats + (size_t)idx * 16) = *reinterpret_cast<v16h*>(tmp);
}

// ---------------------------------------------------------------------------
// Weight repack kernels: emit per-fragment layout [frag][lane][16 halfs] where
// lane<16 holds K=frag_k+0..15 for N=lane, lane>=16 holds K=frag_k+16..31 for
// N=lane-16 (contiguous K halves, same order as this file's A loads).
// ---------------------------------------------------------------------------
__global__ void pack_w1_kernel(const float* __restrict__ w1, _Float16* __restrict__ w1p)
{
    int idx = blockIdx.x * blockDim.x + threadIdx.x;
    if (idx >= W1P_HALFS) return;
    int j = idx & 15, lane = (idx >> 4) & 31, f = idx >> 9;
    int nb = f & 1, ci = f >> 1;
    int k   = ci * 32 + ((lane >> 4) << 4) + j;        // padded K (160)
    int nch = nb * 16 + (lane & 15);                   // out channel 0..31
    _Float16 val = (_Float16)0.f;
    if (k < 144) {
        int tap = k >> 4, c = k & 15;                  // tap-major, 16-padded Cin
        if (c < 12)
            val = (_Float16)w1[(((size_t)nch * 12 + c) * 3 + tap / 3) * 3 + tap % 3];
    }
    w1p[idx] = val;
}

__global__ void pack_w2_kernel(const float* __restrict__ w2, _Float16* __restrict__ w2p)
{
    int idx = blockIdx.x * blockDim.x + threadIdx.x;
    if (idx >= W2P_HALFS) return;
    int j = idx & 15, lane = (idx >> 4) & 31, f = idx >> 9;
    int nb = f & 3, tap = f >> 2;
    int cin = ((lane >> 4) << 4) + j;                  // 0..31
    int nch = nb * 16 + (lane & 15);                   // 0..63
    w2p[idx] = (_Float16)w2[(((size_t)nch * 32 + cin) * 3 + tap / 3) * 3 + tap % 3];
}

__global__ void pack_w3_kernel(const float* __restrict__ w3, _Float16* __restrict__ w3p)
{
    int idx = blockIdx.x * blockDim.x + threadIdx.x;
    if (idx >= W3P_HALFS) return;
    int j = idx & 15, lane = (idx >> 4) & 31, ck = idx >> 9;
    int cin = ck * 32 + ((lane >> 4) << 4) + j;        // 0..63
    int nch = lane & 15;                               // 8 real + 8 pad cols
    w3p[idx] = (nch < 8) ? (_Float16)w3[(size_t)nch * 64 + cin] : (_Float16)0.f;
}

// Cooperative global->LDS copy in 16B units, then block barrier.
__device__ __forceinline__ void stage_weights(_Float16* dst, const _Float16* src, int halfs)
{
    const int vecs = halfs >> 3;                       // 16B chunks
    const uint4* s = reinterpret_cast<const uint4*>(src);
    uint4*       d = reinterpret_cast<uint4*>(dst);
    for (int i = threadIdx.x; i < vecs; i += blockDim.x) d[i] = s[i];
    __syncthreads();
}

// ---------------------------------------------------------------------------
// conv1: 3x3, 16(pad of 12)->32, WMMA implicit GEMM, fused BN+ReLU.
// Each wave: 16 pixels x 32 outch. 5 K-chunks of 32 (taps paired, K padded).
// ---------------------------------------------------------------------------
template <bool GUARD>
__device__ __forceinline__ void conv1_body(
    const _Float16* __restrict__ feats, const _Float16* __restrict__ sW,
    int n, int y, int x0, int mIdx, int hiHalf, int lane, v8f& c0, v8f& c1)
{
#pragma unroll
    for (int ci = 0; ci < 5; ++ci) {
        int tap = ci * 2 + hiHalf;                     // K half selects tap
        int ky = tap / 3 - 1, kx = tap % 3 - 1;
        int yy = y + ky, xx = x0 + mIdx + kx;
        v16h a = {};
        bool ok = (ci < 4) || (tap < 9);               // chunk 4 hi-half is K pad
        if (GUARD) ok = ok && yy >= 0 && yy < HH && xx >= 0 && xx < WW;
        if (ok)
            a = *reinterpret_cast<const v16h*>(
                    feats + ((size_t)((n * HH + yy) * WW + xx)) * 16);
        v16h b0 = *reinterpret_cast<const v16h*>(sW + ((ci * 2 + 0) * 32 + lane) * 16);
        v16h b1 = *reinterpret_cast<const v16h*>(sW + ((ci * 2 + 1) * 32 + lane) * 16);
        c0 = __builtin_amdgcn_wmma_f32_16x16x32_f16(false, a, false, b0, (short)0, c0, false, false);
        c1 = __builtin_amdgcn_wmma_f32_16x16x32_f16(false, a, false, b1, (short)0, c1, false, false);
    }
}

__global__ __launch_bounds__(256) void conv1_wmma_kernel(
    const _Float16* __restrict__ feats,  // [N,H,W,16]
    const _Float16* __restrict__ w1p,    // [5*2][32][16]
    const float* __restrict__ g, const float* __restrict__ bb,
    const float* __restrict__ mm, const float* __restrict__ vv,
    _Float16* __restrict__ out1)         // [N,H,W,32]
{
    __shared__ _Float16 sW[W1P_HALFS];
    stage_weights(sW, w1p, W1P_HALFS);

    const int lane   = threadIdx.x & 31;
    const int wave   = threadIdx.x >> 5;
    const int p0     = (blockIdx.x * 8 + wave) * 16;
    const int n      = p0 / (HH * WW);
    const int rem    = p0 - n * HH * WW;
    const int y      = rem / WW;
    const int x0     = rem - y * WW;
    const int mIdx   = lane & 15;
    const int hiHalf = lane >> 4;

    v8f c0 = {}; v8f c1 = {};
    // Wave-uniform interior test: halo spans y-1..y+1, x0-1..x0+16.
    if (y >= 1 && y <= HH - 2 && x0 >= 1 && x0 + 16 <= WW - 1)
        conv1_body<false>(feats, sW, n, y, x0, mIdx, hiHalf, lane, c0, c1);
    else
        conv1_body<true >(feats, sW, n, y, x0, mIdx, hiHalf, lane, c0, c1);

    const int ch0 = mIdx, ch1 = 16 + mIdx;
    float inv0 = g[ch0] * rsqrtf(vv[ch0] + BN_EPS);
    float bi0  = bb[ch0] - mm[ch0] * inv0;
    float inv1 = g[ch1] * rsqrtf(vv[ch1] + BN_EPS);
    float bi1  = bb[ch1] - mm[ch1] * inv1;
    size_t rowBase = (size_t)(n * HH + y) * WW;
#pragma unroll
    for (int r = 0; r < 8; ++r) {
        int    mp = r + hiHalf * 8;                    // C/D layout row
        size_t pb = (rowBase + x0 + mp) * 32;
        float f0 = c0[r] * inv0 + bi0; f0 = f0 > 0.f ? f0 : 0.f;
        float f1 = c1[r] * inv1 + bi1; f1 = f1 > 0.f ? f1 : 0.f;
        out1[pb + ch0] = (_Float16)f0;
        out1[pb + ch1] = (_Float16)f1;
    }
}

// ---------------------------------------------------------------------------
// conv2: 3x3, 32->64, the dominant GEMM (19.4 GFLOP). Each wave: 16 pixels x
// 64 outch (A fragment reused 4x). 9 K-chunks (one tap each). B fragments
// resident in LDS; interior waves take the branch-free path.
// ---------------------------------------------------------------------------
template <bool GUARD>
__device__ __forceinline__ void conv2_body(
    const _Float16* __restrict__ in, const _Float16* __restrict__ sW,
    int n, int y, int x0, int mIdx, int hiHalf, int lane,
    v8f& c0, v8f& c1, v8f& c2, v8f& c3)
{
    const int cbase = hiHalf * 16;                     // K half: Cin 0-15 / 16-31
#pragma unroll
    for (int tap = 0; tap < 9; ++tap) {
        int  ky = tap / 3 - 1, kx = tap % 3 - 1;
        int  yy = y + ky, xx = x0 + mIdx + kx;
        v16h a;
        if (GUARD) {
            v16h z = {};
            a = z;
            if (yy >= 0 && yy < HH && xx >= 0 && xx < WW)
                a = *reinterpret_cast<const v16h*>(
                        in + ((size_t)((n * HH + yy) * WW + xx)) * 32 + cbase);
        } else {
            const _Float16* ap = in + ((size_t)((n * HH + yy) * WW + xx)) * 32 + cbase;
            a = *reinterpret_cast<const v16h*>(ap);
            if (tap < 8)   // warm GL2/L0 for next tap row while WMMAs run
                __builtin_prefetch(ap + 32, 0, 1);
        }
        const _Float16* wp = sW + (tap * 4) * 32 * 16 + lane * 16;
        v16h b0 = *reinterpret_cast<const v16h*>(wp + 0 * 32 * 16);
        v16h b1 = *reinterpret_cast<const v16h*>(wp + 1 * 32 * 16);
        v16h b2 = *reinterpret_cast<const v16h*>(wp + 2 * 32 * 16);
        v16h b3 = *reinterpret_cast<const v16h*>(wp + 3 * 32 * 16);
        c0 = __builtin_amdgcn_wmma_f32_16x16x32_f16(false, a, false, b0, (short)0, c0, false, false);
        c1 = __builtin_amdgcn_wmma_f32_16x16x32_f16(false, a, false, b1, (short)0, c1, false, false);
        c2 = __builtin_amdgcn_wmma_f32_16x16x32_f16(false, a, false, b2, (short)0, c2, false, false);
        c3 = __builtin_amdgcn_wmma_f32_16x16x32_f16(false, a, false, b3, (short)0, c3, false, false);
    }
}

__global__ __launch_bounds__(256) void conv2_wmma_kernel(
    const _Float16* __restrict__ in,     // [N,H,W,32]
    const _Float16* __restrict__ w2p,    // [9*4][32][16]
    const float* __restrict__ g, const float* __restrict__ bb,
    const float* __restrict__ mm, const float* __restrict__ vv,
    _Float16* __restrict__ out2)         // [N,H,W,64]
{
    __shared__ _Float16 sW[W2P_HALFS];
    stage_weights(sW, w2p, W2P_HALFS);

    const int lane   = threadIdx.x & 31;
    const int wave   = threadIdx.x >> 5;
    const int p0     = (blockIdx.x * 8 + wave) * 16;
    const int n      = p0 / (HH * WW);
    const int rem    = p0 - n * HH * WW;
    const int y      = rem / WW;
    const int x0     = rem - y * WW;
    const int mIdx   = lane & 15;
    const int hiHalf = lane >> 4;

    v8f c0 = {}, c1 = {}, c2 = {}, c3 = {};
    if (y >= 1 && y <= HH - 2 && x0 >= 1 && x0 + 16 <= WW - 1)
        conv2_body<false>(in, sW, n, y, x0, mIdx, hiHalf, lane, c0, c1, c2, c3);
    else
        conv2_body<true >(in, sW, n, y, x0, mIdx, hiHalf, lane, c0, c1, c2, c3);

    size_t rowBase = (size_t)(n * HH + y) * WW;
#pragma unroll
    for (int nb = 0; nb < 4; ++nb) {
        const v8f acc = (nb == 0) ? c0 : (nb == 1) ? c1 : (nb == 2) ? c2 : c3;
        int   ch  = nb * 16 + mIdx;
        float inv = g[ch] * rsqrtf(vv[ch] + BN_EPS);
        float bia = bb[ch] - mm[ch] * inv;
#pragma unroll
        for (int r = 0; r < 8; ++r) {
            int    mp = r + hiHalf * 8;
            size_t pb = (rowBase + x0 + mp) * 64;
            float f = acc[r] * inv + bia; f = f > 0.f ? f : 0.f;
            out2[pb + ch] = (_Float16)f;
        }
    }
}

// ---------------------------------------------------------------------------
// conv3: 1x1, 64->8 (+ReLU) as a single 16x16 WMMA tile per wave (8 real
// columns, 8 zero-padded), K = 64 = 2 chunks. No spatial halo -> no guards.
// ---------------------------------------------------------------------------
__global__ __launch_bounds__(256) void conv3_wmma_kernel(
    const _Float16* __restrict__ in,     // [N,H,W,64]
    const _Float16* __restrict__ w3p,    // [2][32][16]
    float* __restrict__ aff)             // [N,H,W,8]
{
    __shared__ _Float16 sW[W3P_HALFS];
    stage_weights(sW, w3p, W3P_HALFS);

    const int lane   = threadIdx.x & 31;
    const int wave   = threadIdx.x >> 5;
    const int p0     = (blockIdx.x * 8 + wave) * 16;
    const int n      = p0 / (HH * WW);
    const int rem    = p0 - n * HH * WW;
    const int y      = rem / WW;
    const int x0     = rem - y * WW;
    const int mIdx   = lane & 15;
    const int hiHalf = lane >> 4;

    v8f c = {};
#pragma unroll
    for (int ck = 0; ck < 2; ++ck) {
        const _Float16* ap = in + ((size_t)(p0 + mIdx)) * 64 + ck * 32 + hiHalf * 16;
        v16h a = *reinterpret_cast<const v16h*>(ap);
        v16h b = *reinterpret_cast<const v16h*>(sW + (ck * 32 + lane) * 16);
        c = __builtin_amdgcn_wmma_f32_16x16x32_f16(false, a, false, b, (short)0, c, false, false);
    }
    if (mIdx < 8) {
        size_t rowBase = (size_t)(n * HH + y) * WW;
#pragma unroll
        for (int r = 0; r < 8; ++r) {
            int   mp = r + hiHalf * 8;
            float f  = c[r]; f = f > 0.f ? f : 0.f;
            aff[(rowBase + x0 + mp) * 8 + mIdx] = f;
        }
    }
}

// ---------------------------------------------------------------------------
// Kernel 5: 8-neighbor softmax gate propagation + blend.
// gate channel k samples aff[k] and disp at neighbor offset; OOB contributes
// exp(0) to the softmax denominator (matching jnp.pad-with-zeros semantics).
// ---------------------------------------------------------------------------
__global__ __launch_bounds__(256) void propagate_kernel(
    const float* __restrict__ aff,       // [N,H,W,8]
    const float* __restrict__ disp,      // [N,1,H,W]
    float* __restrict__ out)             // [N,1,H,W]
{
    int idx = blockIdx.x * blockDim.x + threadIdx.x;
    if (idx >= NPIX) return;
    int n   = idx / (HH * WW);
    int rem = idx - n * HH * WW;
    int y   = rem / WW;
    int x   = rem - y * WW;

    const int offy[8] = { 1, 1, 1, 0, 0, -1, -1, -1 };
    const int offx[8] = { 1, 0, -1, 1, -1, 1, 0, -1 };

    float vals[8], dv[8];
#pragma unroll
    for (int k = 0; k < 8; ++k) {
        int yy = y + offy[k], xx = x + offx[k];
        bool ok = (yy >= 0) && (yy < HH) && (xx >= 0) && (xx < WW);
        size_t p = (size_t)(n * HH + yy) * WW + xx;
        vals[k] = ok ? aff[p * 8 + k] : 0.f;
        dv[k]   = ok ? disp[p]        : 0.f;
    }
    float mx = vals[0];
#pragma unroll
    for (int k = 1; k < 8; ++k) mx = fmaxf(mx, vals[k]);
    float s = 0.f, acc = 0.f;
#pragma unroll
    for (int k = 0; k < 8; ++k) {
        float e = expf(vals[k] - mx);
        s += e; acc += e * dv[k];
    }
    out[idx] = 0.3f * (acc / s) + 0.7f * disp[idx];
}

// ---------------------------------------------------------------------------
extern "C" void kernel_launch(void* const* d_in, const int* in_sizes, int n_in,
                              void* d_out, int out_size, void* d_ws, size_t ws_size,
                              hipStream_t stream) {
    const float* normal = (const float*)d_in[0];
    const float* left   = (const float*)d_in[1];
    const float* right  = (const float*)d_in[2];
    const float* disp   = (const float*)d_in[3];
    const float* w1     = (const float*)d_in[4];
    const float* g1     = (const float*)d_in[5];
    const float* b1     = (const float*)d_in[6];
    const float* m1     = (const float*)d_in[7];
    const float* v1     = (const float*)d_in[8];
    const float* w2     = (const float*)d_in[9];
    const float* g2     = (const float*)d_in[10];
    const float* b2     = (const float*)d_in[11];
    const float* m2     = (const float*)d_in[12];
    const float* v2     = (const float*)d_in[13];
    const float* w3     = (const float*)d_in[14];
    float* out = (float*)d_out;

    // Workspace layout (all offsets 256B aligned).
    char* ws = (char*)d_ws;
    size_t off = 0;
    _Float16* feats = (_Float16*)(ws + off); off += (size_t)NPIX * 16 * sizeof(_Float16); // 16 MB
    _Float16* o1    = (_Float16*)(ws + off); off += (size_t)NPIX * 32 * sizeof(_Float16); // 32 MB
    _Float16* o2    = (_Float16*)(ws + off); off += (size_t)NPIX * 64 * sizeof(_Float16); // 64 MB
    float*    aff   = (float*)   (ws + off); off += (size_t)NPIX * 8  * sizeof(float);    // 16 MB
    _Float16* w1p   = (_Float16*)(ws + off); off += W1P_HALFS * sizeof(_Float16);
    _Float16* w2p   = (_Float16*)(ws + off); off += W2P_HALFS * sizeof(_Float16);
    _Float16* w3p   = (_Float16*)(ws + off); off += W3P_HALFS * sizeof(_Float16);
    (void)ws_size; (void)in_sizes; (void)n_in; (void)out_size;

    // Weight repacks (tiny).
    pack_w1_kernel<<<(W1P_HALFS + 255) / 256, 256, 0, stream>>>(w1, w1p);
    pack_w2_kernel<<<(W2P_HALFS + 255) / 256, 256, 0, stream>>>(w2, w2p);
    pack_w3_kernel<<<(W3P_HALFS + 255) / 256, 256, 0, stream>>>(w3, w3p);

    // Feature prep.
    prep_feats_kernel<<<NPIX / 256, 256, 0, stream>>>(normal, left, right, disp, feats);

    // WMMA conv stack: 8 waves/block, 16 pixels/wave -> NPIX/128 blocks.
    const int convBlocks = NPIX / (8 * 16);
    conv1_wmma_kernel<<<convBlocks, 256, 0, stream>>>(feats, w1p, g1, b1, m1, v1, o1);
    conv2_wmma_kernel<<<convBlocks, 256, 0, stream>>>(o1, w2p, g2, b2, m2, v2, o2);
    conv3_wmma_kernel<<<convBlocks, 256, 0, stream>>>(o2, w3p, aff);

    // Softmax propagation + blend.
    propagate_kernel<<<NPIX / 256, 256, 0, stream>>>(aff, disp, out);
}
